// TATLayer_16492674417483
// MI455X (gfx1250) — matrix operand order repaired and checked by reading, not compiled, not run on hardware
//
#include <hip/hip_runtime.h>
#include <math.h>

// ---------------------------------------------------------------------------
// Types for CDNA5 WMMA
// ---------------------------------------------------------------------------
typedef __attribute__((ext_vector_type(16))) __bf16 v16bf;
typedef __attribute__((ext_vector_type(8)))  __bf16 v8bf;
typedef __attribute__((ext_vector_type(8)))  float  v8f;

union UA { v16bf v; v8bf h[2]; };

#define C_DIM   256
#define HEADS   8
#define DHEAD   32
#define NWIN    134
#define LQ      128
#define LV      192
#define LOGMAXF 4.6051701859880914f   // log(1/0.01)

// ===========================================================================
// Weight prep: Wt[n][k] = (bf16) W[k][n]   (256x256, tiled transpose)
// grid = 64 blocks (8x8 tiles of 32x32), 256 threads
// ===========================================================================
__global__ __launch_bounds__(256) void wt_prep_kernel(
    const float* __restrict__ W, __bf16* __restrict__ Wt)
{
  __shared__ float T[32][33];
  const int bx = blockIdx.x & 7;        // k tile
  const int by = blockIdx.x >> 3;       // n tile
  const int k0 = bx * 32, n0 = by * 32;
  const int tr = threadIdx.x >> 5, tc = threadIdx.x & 31;
  #pragma unroll
  for (int i = 0; i < 4; ++i)
    T[tr + i * 8][tc] = W[(size_t)(k0 + tr + i * 8) * C_DIM + n0 + tc];
  __syncthreads();
  #pragma unroll
  for (int i = 0; i < 4; ++i)
    Wt[(size_t)(n0 + tr + i * 8) * C_DIM + k0 + tc] = (__bf16)T[tc][tr + i * 8];
}

// ===========================================================================
// GEMM: Y[M x 256] = act( X[M x 256] @ W[256 x 256] + bias ), W given as
// bf16 transposed Wt[n][k]. Block: 256 threads (8 waves), tile 64(M) x 128(N).
// Each wave computes 4 output tiles (1 m-tile x 4 n-tiles): one A fragment
// per K-step feeds 4 WMMAs. X tile staged once in LDS; single barrier.
// M must be a multiple of 64.
// ===========================================================================
__global__ __launch_bounds__(256) void gemm_mlp_kernel(
    const float* __restrict__ X, const __bf16* __restrict__ Wt,
    const float* __restrict__ bias, float* __restrict__ Y,
    int M, int act)
{
  __shared__ __bf16 Xs[64][264];   // 256 + 8 pad -> row stride 528B (16B mult)

  const int tid  = threadIdx.x;
  const int lane = tid & 31;
  const int wave = tid >> 5;
  const int col  = lane & 15;
  const int half = lane >> 4;

  const int m0 = blockIdx.y * 64;
  const int n0 = blockIdx.x * 128;
  const int mi     = wave & 3;
  const int njBase = (wave >> 2) * 4;

  // stage whole X tile (64 x 256) as bf16, vectorized
  {
    const int r0 = tid >> 5;          // 0..7
    const int c  = (tid & 31) * 8;    // 0..248
    #pragma unroll
    for (int i = 0; i < 8; ++i) {
      const int r = i * 8 + r0;
      const float4* xp = (const float4*)(X + (size_t)(m0 + r) * C_DIM + c);
      float4 a0 = xp[0], a1 = xp[1];
      v8bf pk;
      pk[0] = (__bf16)a0.x; pk[1] = (__bf16)a0.y;
      pk[2] = (__bf16)a0.z; pk[3] = (__bf16)a0.w;
      pk[4] = (__bf16)a1.x; pk[5] = (__bf16)a1.y;
      pk[6] = (__bf16)a1.z; pk[7] = (__bf16)a1.w;
      *(v8bf*)&Xs[r][c] = pk;
    }
  }
  __syncthreads();

  const int lrow = mi * 16 + col;
  const __bf16* wr[4];
  int ng[4];
  #pragma unroll
  for (int j = 0; j < 4; ++j) {
    ng[j] = n0 + (njBase + j) * 16 + col;
    wr[j] = Wt + (size_t)ng[j] * C_DIM;
  }

  v8f acc[4] = {{}, {}, {}, {}};
  #pragma unroll
  for (int k0 = 0; k0 < C_DIM; k0 += 32) {
    UA a;
    a.h[0] = *(const v8bf*)&Xs[lrow][k0 + half * 16];
    a.h[1] = *(const v8bf*)&Xs[lrow][k0 + half * 16 + 8];
    UA b[4];
    #pragma unroll
    for (int j = 0; j < 4; ++j) {
      b[j].h[0] = *(const v8bf*)(wr[j] + k0 + half * 16);
      b[j].h[1] = *(const v8bf*)(wr[j] + k0 + half * 16 + 8);
    }
    #pragma unroll
    for (int j = 0; j < 4; ++j)
      acc[j] = __builtin_amdgcn_wmma_f32_16x16x32_bf16(
          false, a.v, false, b[j].v, (short)0, acc[j], false, false);
  }

  // epilogue: bias (+ SiLU). C/D layout: m = r + 8*half, n = col
  float bv[4];
  #pragma unroll
  for (int j = 0; j < 4; ++j) bv[j] = bias[ng[j]];
  #pragma unroll
  for (int r = 0; r < 8; ++r) {
    const int m = m0 + mi * 16 + r + 8 * half;
    float* yp = Y + (size_t)m * C_DIM;
    #pragma unroll
    for (int j = 0; j < 4; ++j) {
      float y = acc[j][r] + bv[j];
      if (act) y = y / (1.f + __expf(-y));
      yp[ng[j]] = y;
    }
  }
}

// ===========================================================================
// Gather: pyramid -> windowed bf16 [B, NWIN, Lw, 256]  (c fastest)
// pad (zeros) + roll(+4,+4); subLast: v = p[:, t] - p[:, Tsrc-1]
// ===========================================================================
__global__ void gather_win_kernel(
    const float* __restrict__ s0, const float* __restrict__ s1,
    const float* __restrict__ s2, __bf16* __restrict__ out,
    int Lw, int Tsrc, int subLast, size_t N)
{
  size_t idx = (size_t)blockIdx.x * 256 + threadIdx.x;
  if (idx >= N) return;
  int c = (int)(idx & 255);
  size_t rr = idx >> 8;
  int l   = (int)(rr % Lw); rr /= Lw;
  int win = (int)(rr % NWIN);
  int b   = (int)(rr / NWIN);

  const float* src; int H, nW, off;
  if (win < 100)      { src = s0; H = 80; nW = 10; off = 0;   }
  else if (win < 125) { src = s1; H = 40; nW = 5;  off = 100; }
  else                { src = s2; H = 20; nW = 3;  off = 125; }

  int wl = win - off;
  int wh = wl / nW, ww = wl % nW;
  int t  = l >> 6;
  int rem = l & 63;
  int dh = rem >> 3, dw = rem & 7;
  int Hp = nW * 8;
  int sh = wh * 8 + dh - 4; if (sh < 0) sh += Hp;
  int sw = ww * 8 + dw - 4; if (sw < 0) sw += Hp;

  float v = 0.f;
  if (sh < H && sw < H) {
    v = src[(((size_t)(b * Tsrc + t) * H + sh) * H + sw) * C_DIM + c];
    if (subLast)
      v -= src[(((size_t)(b * Tsrc + (Tsrc - 1)) * H + sh) * H + sw) * C_DIM + c];
  }
  out[idx] = (__bf16)v;
}

// ===========================================================================
// Gather (transposed): v windows as Vt[B, NWIN, 256, LV]  (l fastest)
// ===========================================================================
__global__ void gather_vt_kernel(
    const float* __restrict__ s0, const float* __restrict__ s1,
    const float* __restrict__ s2, __bf16* __restrict__ out, size_t N)
{
  size_t idx = (size_t)blockIdx.x * 256 + threadIdx.x;
  if (idx >= N) return;
  int l = (int)(idx % LV);
  size_t rr = idx / LV;
  int c   = (int)(rr & 255); rr >>= 8;
  int win = (int)(rr % NWIN);
  int b   = (int)(rr / NWIN);

  const float* src; int H, nW, off;
  if (win < 100)      { src = s0; H = 80; nW = 10; off = 0;   }
  else if (win < 125) { src = s1; H = 40; nW = 5;  off = 100; }
  else                { src = s2; H = 20; nW = 3;  off = 125; }

  int wl = win - off;
  int wh = wl / nW, ww = wl % nW;
  int t  = l >> 6;
  int rem = l & 63;
  int dh = rem >> 3, dw = rem & 7;
  int Hp = nW * 8;
  int sh = wh * 8 + dh - 4; if (sh < 0) sh += Hp;
  int sw = ww * 8 + dw - 4; if (sw < 0) sw += Hp;

  float v = 0.f;
  if (sh < H && sw < H) {
    v = src[(((size_t)(b * 4 + t) * H + sh) * H + sw) * C_DIM + c]
      - src[(((size_t)(b * 4 + 3) * H + sh) * H + sw) * C_DIM + c];
  }
  out[idx] = (__bf16)v;
}

// ===========================================================================
// Window attention. Block = (b,window), 8 waves = 8 heads.
// S = QK^T (12 WMMA, K=32) -> softmax(+pos, *scale) -> P via LDS -> P@V.
// ===========================================================================
__global__ __launch_bounds__(256) void attn_kernel(
    const __bf16* __restrict__ Q, const __bf16* __restrict__ K,
    const __bf16* __restrict__ Vt, const float* __restrict__ pos,
    const float* __restrict__ lscale, float* __restrict__ O)
{
  __shared__ __bf16 Pl[8][16][LV + 8];   // row stride 400B (16B multiple)

  const int bw   = blockIdx.x;           // b*NWIN + win
  const int b    = bw / NWIN;
  const int wave = threadIdx.x >> 5;     // head
  const int lane = threadIdx.x & 31;
  const int col  = lane & 15;
  const int half = lane >> 4;
  const int head = wave;

  const float scale = __expf(fminf(lscale[head], LOGMAXF));

  const __bf16* qb  = Q  + (size_t)bw * LQ * C_DIM + head * DHEAD;
  const __bf16* kpb = K  + (size_t)bw * LV * C_DIM + head * DHEAD;
  const __bf16* vtb = Vt + ((size_t)bw * C_DIM + head * DHEAD) * LV;
  const float*  pb  = pos + (size_t)(b * HEADS + head) * LQ * LV;
  float*        ob  = O + (size_t)bw * LQ * C_DIM + head * DHEAD;

  for (int mt = 0; mt < 8; ++mt) {
    // q A-fragment
    const __bf16* qp = qb + (size_t)(mt * 16 + col) * C_DIM;
    UA aq;
    aq.h[0] = *(const v8bf*)(qp + half * 16);
    aq.h[1] = *(const v8bf*)(qp + half * 16 + 8);

    // S = q @ k^T : 12 tiles of 16x16, single K=32 WMMA each
    v8f S[12];
    #pragma unroll
    for (int t = 0; t < 12; ++t) {
      const __bf16* kp = kpb + (size_t)(t * 16 + col) * C_DIM + half * 16;
      UA bk;
      bk.h[0] = *(const v8bf*)kp;
      bk.h[1] = *(const v8bf*)(kp + 8);
      v8f z = {};
      S[t] = __builtin_amdgcn_wmma_f32_16x16x32_bf16(false, aq.v, false, bk.v, (short)0, z, false, false);
    }

    // softmax over 192 cols per row (rows: m = r + 8*half)
    #pragma unroll
    for (int r = 0; r < 8; ++r) {
      const int mg = mt * 16 + r + 8 * half;
      float mx = -3.4e38f;
      #pragma unroll
      for (int t = 0; t < 12; ++t) {
        float v = S[t][r] * scale + pb[(size_t)mg * LV + t * 16 + col];
        S[t][r] = v;
        mx = fmaxf(mx, v);
      }
      #pragma unroll
      for (int d = 1; d < 16; d <<= 1) mx = fmaxf(mx, __shfl_xor(mx, d, 32));
      float sum = 0.f;
      #pragma unroll
      for (int t = 0; t < 12; ++t) {
        float e = __expf(S[t][r] - mx);
        S[t][r] = e;
        sum += e;
      }
      #pragma unroll
      for (int d = 1; d < 16; d <<= 1) sum += __shfl_xor(sum, d, 32);
      const float inv = 1.f / sum;
      const int mrow = r + 8 * half;
      #pragma unroll
      for (int t = 0; t < 12; ++t)
        Pl[wave][mrow][t * 16 + col] = (__bf16)(S[t][r] * inv);
    }
    // per-wave LDS region; DS ops are in-order within a wave -> no barrier

    // O = P @ V : out 16 x 32 (2 n-tiles), K=192 in 6 WMMA steps
    v8f O0 = {}, O1 = {};
    #pragma unroll
    for (int s = 0; s < 6; ++s) {
      UA ap;
      ap.h[0] = *(const v8bf*)&Pl[wave][col][s * 32 + half * 16];
      ap.h[1] = *(const v8bf*)&Pl[wave][col][s * 32 + half * 16 + 8];
      const __bf16* v0 = vtb + (size_t)col * LV + s * 32 + half * 16;
      const __bf16* v1 = vtb + (size_t)(col + 16) * LV + s * 32 + half * 16;
      UA bv0, bv1;
      bv0.h[0] = *(const v8bf*)v0;
      bv0.h[1] = *(const v8bf*)(v0 + 8);
      bv1.h[0] = *(const v8bf*)v1;
      bv1.h[1] = *(const v8bf*)(v1 + 8);
      O0 = __builtin_amdgcn_wmma_f32_16x16x32_bf16(false, ap.v, false, bv0.v, (short)0, O0, false, false);
      O1 = __builtin_amdgcn_wmma_f32_16x16x32_bf16(false, ap.v, false, bv1.v, (short)0, O1, false, false);
    }

    #pragma unroll
    for (int r = 0; r < 8; ++r) {
      const int mg = mt * 16 + r + 8 * half;
      ob[(size_t)mg * C_DIM + col]      = O0[r];
      ob[(size_t)mg * C_DIM + 16 + col] = O1[r];
    }
  }
}

// ===========================================================================
// Scatter: windows [B, NWIN, LQ, 256] -> level (B, TF, H, W, C)
// ===========================================================================
__global__ void scatter_win_kernel(
    const float* __restrict__ a, float* __restrict__ fa,
    int H, int nW, int winOff, size_t N)
{
  size_t idx = (size_t)blockIdx.x * 256 + threadIdx.x;
  if (idx >= N) return;
  int c = (int)(idx & 255);
  size_t rr = idx >> 8;
  int w = (int)(rr % H); rr /= H;
  int h = (int)(rr % H); rr /= H;
  int t = (int)(rr % 2);
  int b = (int)(rr / 2);
  int Hp = nW * 8;
  int hp = h + 4; if (hp >= Hp) hp -= Hp;
  int wp = w + 4; if (wp >= Hp) wp -= Hp;
  int win = winOff + (hp >> 3) * nW + (wp >> 3);
  int l = t * 64 + (hp & 7) * 8 + (wp & 7);
  fa[idx] = a[((size_t)(b * NWIN + win) * LQ + l) * C_DIM + c];
}

// ===========================================================================
// out = res + LayerNorm(x) * g + b   (one wave per token, C=256)
// ===========================================================================
__global__ __launch_bounds__(256) void ln_res_kernel(
    const float* __restrict__ x, const float* __restrict__ res,
    const float* __restrict__ g, const float* __restrict__ bb,
    float* __restrict__ out, int ntok)
{
  const int wave = threadIdx.x >> 5;
  const int lane = threadIdx.x & 31;
  const int tok = blockIdx.x * 8 + wave;
  if (tok >= ntok) return;
  const float* xp = x + (size_t)tok * C_DIM;
  float v[8];
  float s = 0.f, s2 = 0.f;
  #pragma unroll
  for (int j = 0; j < 8; ++j) {
    v[j] = xp[lane + 32 * j];
    s  += v[j];
    s2 += v[j] * v[j];
  }
  #pragma unroll
  for (int d = 1; d < 32; d <<= 1) {
    s  += __shfl_xor(s,  d, 32);
    s2 += __shfl_xor(s2, d, 32);
  }
  const float mean = s * (1.f / 256.f);
  const float var  = s2 * (1.f / 256.f) - mean * mean;
  const float inv  = rsqrtf(var + 1e-5f);
  const float* rp = res + (size_t)tok * C_DIM;
  float* op = out + (size_t)tok * C_DIM;
  #pragma unroll
  for (int j = 0; j < 8; ++j) {
    int c = lane + 32 * j;
    op[c] = rp[c] + (v[j] - mean) * inv * g[c] + bb[c];
  }
}

// ===========================================================================
// Host orchestration
// ===========================================================================
extern "C" void kernel_launch(void* const* d_in, const int* in_sizes, int n_in,
                              void* d_out, int out_size, void* d_ws, size_t ws_size,
                              hipStream_t stream) {
  (void)in_sizes; (void)n_in; (void)out_size; (void)ws_size;

  const float* fp[3] = {(const float*)d_in[0], (const float*)d_in[1], (const float*)d_in[2]};
  const float* ff[3] = {(const float*)d_in[3], (const float*)d_in[4], (const float*)d_in[5]};
  const float* pos        = (const float*)d_in[6];
  const float* fc_in_w1   = (const float*)d_in[7];
  const float* fc_in_b1   = (const float*)d_in[8];
  const float* fc_in_w2   = (const float*)d_in[9];
  const float* fc_in_b2   = (const float*)d_in[10];
  const float* fc_out_w1  = (const float*)d_in[11];
  const float* fc_out_b1  = (const float*)d_in[12];
  const float* fc_out_w2  = (const float*)d_in[13];
  const float* fc_out_b2  = (const float*)d_in[14];
  const float* mlp_w1     = (const float*)d_in[15];
  const float* mlp_b1     = (const float*)d_in[16];
  const float* mlp_w2     = (const float*)d_in[17];
  const float* mlp_b2     = (const float*)d_in[18];
  const float* norm1_g    = (const float*)d_in[19];
  const float* norm1_b    = (const float*)d_in[20];
  const float* norm2_g    = (const float*)d_in[21];
  const float* norm2_b    = (const float*)d_in[22];
  const float* lscale     = (const float*)d_in[23];
  float* outp = (float*)d_out;

  static const int Hs[3]   = {80, 40, 20};
  static const int nWs[3]  = {10, 5, 3};
  static const int wOff[3] = {0, 100, 125};
  int tokP[3], tokF[3];
  size_t offP[3], offF[3], outOff[3];
  size_t accP = 0, accF = 0;
  for (int i = 0; i < 3; ++i) {
    tokP[i] = 2 * 4 * Hs[i] * Hs[i];
    tokF[i] = 2 * 2 * Hs[i] * Hs[i];
    offP[i] = accP; accP += (size_t)tokP[i] * C_DIM;
    offF[i] = accF; outOff[i] = accF; accF += (size_t)tokF[i] * C_DIM;
  }

  // workspace carve-up
  float* Pin  = (float*)d_ws;                    // 17,203,200 f
  float* Fin  = Pin  + accP;                     //  8,601,600 f
  float* Hd   = Fin  + accF;                     // 13,107,200 f
  float* Xb   = Hd   + (size_t)tokP[0] * C_DIM;  //  8,601,600 f
  float* Aout = Xb   + accF;                     //  8,781,824 f
  const size_t nQ = (size_t)2 * NWIN * LQ * C_DIM;
  const size_t nK = (size_t)2 * NWIN * LV * C_DIM;
  __bf16* Qb  = (__bf16*)(Aout + nQ);
  __bf16* Kb  = Qb + nQ;
  __bf16* Vtb = Kb + nK;
  __bf16* WtB = Vtb + nK;                        // 18 x 65536 bf16

  // --- prep transposed bf16 weights ---
  const float* wsrc[6] = {fc_in_w1, fc_in_w2, fc_out_w1, fc_out_w2, mlp_w1, mlp_w2};
  for (int m = 0; m < 6; ++m)
    for (int i = 0; i < 3; ++i)
      wt_prep_kernel<<<64, 256, 0, stream>>>(
          wsrc[m] + (size_t)i * C_DIM * C_DIM,
          WtB + (size_t)(m * 3 + i) * C_DIM * C_DIM);
  auto WT = [&](int m, int i) { return WtB + (size_t)(m * 3 + i) * C_DIM * C_DIM; };

  auto gemm = [&](const float* X, const __bf16* Wt, const float* B,
                  float* Y, int M, int act) {
    dim3 grid(2, (unsigned)(M / 64));
    gemm_mlp_kernel<<<grid, 256, 0, stream>>>(X, Wt, B, Y, M, act);
  };

  // 1) p_in = mlp2(fp) ; f_in = mlp2(ff)   (fc_in weights per level)
  for (int i = 0; i < 3; ++i) {
    const size_t bo = (size_t)i * C_DIM;
    gemm(fp[i], WT(0, i), fc_in_b1 + bo, Hd, tokP[i], 1);
    gemm(Hd,    WT(1, i), fc_in_b2 + bo, Pin + offP[i], tokP[i], 0);
    gemm(ff[i], WT(0, i), fc_in_b1 + bo, Hd, tokF[i], 1);
    gemm(Hd,    WT(1, i), fc_in_b2 + bo, Fin + offF[i], tokF[i], 0);
  }

  // 2) windowize q (from f_in), k (p_in), v transposed (p_in)
  gather_win_kernel<<<(unsigned)(nQ / 256), 256, 0, stream>>>(
      Fin + offF[0], Fin + offF[1], Fin + offF[2], Qb, LQ, 2, 0, nQ);
  gather_win_kernel<<<(unsigned)(nK / 256), 256, 0, stream>>>(
      Pin + offP[0], Pin + offP[1], Pin + offP[2], Kb, LV, 4, 0, nK);
  gather_vt_kernel<<<(unsigned)(nK / 256), 256, 0, stream>>>(
      Pin + offP[0], Pin + offP[1], Pin + offP[2], Vtb, nK);

  // 3) attention
  attn_kernel<<<2 * NWIN, 256, 0, stream>>>(Qb, Kb, Vtb, pos, lscale, Aout);

  // 4) windows -> pyramid (reuse Fin region as fa)
  for (int i = 0; i < 3; ++i) {
    size_t Ni = (size_t)tokF[i] * C_DIM;
    scatter_win_kernel<<<(unsigned)(Ni / 256), 256, 0, stream>>>(
        Aout, Fin + offF[i], Hs[i], nWs[i], wOff[i], Ni);
  }

  // 5) fa = mlp2(fa, fc_out)  (into Pin region, reused)
  for (int i = 0; i < 3; ++i) {
    const size_t bo = (size_t)i * C_DIM;
    gemm(Fin + offF[i], WT(2, i), fc_out_b1 + bo, Hd, tokF[i], 1);
    gemm(Hd,            WT(3, i), fc_out_b2 + bo, Pin + offF[i], tokF[i], 0);
  }

  // 6) x = ff + LN(fa; norm1)
  for (int i = 0; i < 3; ++i)
    ln_res_kernel<<<tokF[i] / 8, 256, 0, stream>>>(
        Pin + offF[i], ff[i], norm1_g + i * C_DIM, norm1_b + i * C_DIM,
        Xb + offF[i], tokF[i]);

  // 7) y2 = mlp2(x, mlp) ; out = x + LN(y2; norm2)
  for (int i = 0; i < 3; ++i) {
    const size_t bo = (size_t)i * C_DIM;
    gemm(Xb + offF[i], WT(4, i), mlp_b1 + bo, Hd, tokF[i], 1);
    gemm(Hd,           WT(5, i), mlp_b2 + bo, Pin + offF[i], tokF[i], 0);
  }
  for (int i = 0; i < 3; ++i)
    ln_res_kernel<<<tokF[i] / 8, 256, 0, stream>>>(
        Pin + offF[i], Xb + offF[i], norm2_g + i * C_DIM, norm2_b + i * C_DIM,
        outp + outOff[i], tokF[i]);
}
